// Fetcher_27556510171455
// MI455X (gfx1250) — matrix operand order repaired and checked by reading, not compiled, not run on hardware
//
#include <hip/hip_runtime.h>
#include <math.h>

typedef unsigned short u16;
typedef unsigned int   u32;
typedef unsigned long long u64;
typedef __attribute__((ext_vector_type(8)))  u32    v8u;
typedef __attribute__((ext_vector_type(16))) __bf16 v16bf;
typedef __attribute__((ext_vector_type(8)))  float  v8f;

#define BDIM   8
#define SEQ    4096
#define DMODEL 768
#define NIMG   1024
#define FFDIM  3072
#define CAP    640
#define HEADS  12
#define DH     64
#define MQ     (BDIM*CAP)    /* 5120 */
#define MI     (BDIM*NIMG)   /* 8192 */
#define SOC_ID 3
#define EOS_ID 2

__device__ __forceinline__ u16 f2bf(float f){
  u32 u = __builtin_bit_cast(u32, f);
  u32 r = (u + 0x7fffu + ((u >> 16) & 1u)) >> 16;
  return (u16)r;
}
__device__ __forceinline__ v16bf asbf(v8u u){ return __builtin_bit_cast(v16bf, u); }

__device__ __forceinline__ float gelu_f(float x){
  float x3 = x*x*x;
  return 0.5f*x*(1.0f + tanhf(0.7978845608028654f*(x + 0.044715f*x3)));
}

// async global->LDS copy of 16B per lane (gfx1250). ldsoff must be 16B aligned.
__device__ __forceinline__ void async_ld_b128(u32 ldsoff, const void* gsrc){
  asm volatile("global_load_async_to_lds_b128 %0, %1, off"
               :: "v"(ldsoff), "v"((u64)gsrc) : "memory");
}
__device__ __forceinline__ void async_wait0(){
  asm volatile("s_wait_asynccnt 0" ::: "memory");
}

// ---------------------------------------------------------------- utilities
__global__ __launch_bounds__(256) void convert_f32_bf16(const float* __restrict__ s,
                                                        u16* __restrict__ d, long n){
  long i = (long)blockIdx.x * blockDim.x + threadIdx.x;
  long stride = (long)gridDim.x * blockDim.x;
  for (; i < n; i += stride) d[i] = f2bf(s[i]);
}

__global__ __launch_bounds__(256) void zero_f32(float* __restrict__ d, long n){
  long i = (long)blockIdx.x * blockDim.x + threadIdx.x;
  long stride = (long)gridDim.x * blockDim.x;
  for (; i < n; i += stride) d[i] = 0.0f;
}

__global__ __launch_bounds__(256) void copy_f32(const float* __restrict__ s,
                                                float* __restrict__ d, long n){
  long i = (long)blockIdx.x * blockDim.x + threadIdx.x;
  long stride = (long)gridDim.x * blockDim.x;
  for (; i < n; i += stride) d[i] = s[i];
}

// ---------------------------------------------------------------- mask scan
__global__ __launch_bounds__(256) void scan_kernel(const int* __restrict__ seq,
                                                   int* __restrict__ pos,
                                                   int* __restrict__ counts){
  const int b = blockIdx.x;
  const int* sq = seq + (size_t)b * SEQ;
  int* pp = pos + (size_t)b * SEQ;
  const int PER = SEQ / 256;
  __shared__ int eosLoc[257], valLoc[257];
  const int t = threadIdx.x;
  const int base = t * PER;
  int e = 0;
  for (int i = 0; i < PER; ++i) e += (sq[base + i] == EOS_ID) ? 1 : 0;
  eosLoc[t + 1] = e;
  if (t == 0) eosLoc[0] = 0;
  __syncthreads();
  if (t == 0) for (int i = 1; i <= 256; ++i) eosLoc[i] += eosLoc[i - 1];
  __syncthreads();
  int epre = eosLoc[t];
  int er = epre, vc = 0;
  for (int i = 0; i < PER; ++i){
    int tok = sq[base + i];
    if (tok == EOS_ID) er++;
    else if (tok == SOC_ID && er == 0) vc++;
  }
  valLoc[t + 1] = vc;
  if (t == 0) valLoc[0] = 0;
  __syncthreads();
  if (t == 0){
    for (int i = 1; i <= 256; ++i) valLoc[i] += valLoc[i - 1];
    int c = valLoc[256];
    counts[b] = c < CAP ? c : CAP;
  }
  __syncthreads();
  int vpre = valLoc[t];
  er = epre;
  for (int i = 0; i < PER; ++i){
    int tok = sq[base + i];
    int p = -1;
    if (tok == EOS_ID) er++;
    else if (tok == SOC_ID && er == 0){ p = vpre; vpre++; }
    pp[base + i] = (p >= 0 && p < CAP) ? p : -1;
  }
}

__global__ __launch_bounds__(256) void extract_kernel(const float* __restrict__ hid,
                                                      const int* __restrict__ pos,
                                                      float* __restrict__ x){
  const int bs = blockIdx.x;
  const int b = bs / SEQ;
  const int p = pos[bs];
  if (p < 0) return;
  const float* src = hid + (size_t)bs * DMODEL;
  float* dst = x + ((size_t)b * CAP + p) * DMODEL;
  for (int c = threadIdx.x; c < DMODEL; c += 256) dst[c] = src[c];
}

__global__ __launch_bounds__(256) void scatter_kernel(const float* __restrict__ hid,
                                                      const float* __restrict__ x,
                                                      const int* __restrict__ pos,
                                                      float* __restrict__ out){
  const int bs = blockIdx.x;
  const int b = bs / SEQ;
  const int p = pos[bs];
  const float* src = (p >= 0) ? (x + ((size_t)b * CAP + p) * DMODEL)
                              : (hid + (size_t)bs * DMODEL);
  float* dst = out + (size_t)bs * DMODEL;
  for (int c = threadIdx.x; c < DMODEL; c += 256) dst[c] = src[c];
}

// ---------------------------------------------------------------- layernorm -> bf16
__global__ __launch_bounds__(256) void ln_kernel(const float* __restrict__ X,
                                                 const float* __restrict__ sc,
                                                 const float* __restrict__ bi,
                                                 u16* __restrict__ out){
  const int row = blockIdx.x;
  const float* xr = X + (size_t)row * DMODEL;
  float a = 0.f, a2 = 0.f;
  for (int c = threadIdx.x; c < DMODEL; c += 256){ float v = xr[c]; a += v; a2 += v*v; }
  __shared__ float s1[256], s2[256];
  s1[threadIdx.x] = a; s2[threadIdx.x] = a2;
  __syncthreads();
  for (int st = 128; st > 0; st >>= 1){
    if (threadIdx.x < st){ s1[threadIdx.x] += s1[threadIdx.x+st]; s2[threadIdx.x] += s2[threadIdx.x+st]; }
    __syncthreads();
  }
  const float mu = s1[0] * (1.0f/DMODEL);
  float var = s2[0] * (1.0f/DMODEL) - mu*mu;
  var = var > 0.f ? var : 0.f;
  const float rs = rsqrtf(var + 1e-5f);
  u16* orow = out + (size_t)row * DMODEL;
  for (int c = threadIdx.x; c < DMODEL; c += 256)
    orow[c] = f2bf((xr[c] - mu) * rs * sc[c] + bi[c]);
}

// ---------------------------------------------------------------- WMMA GEMM
// C[M,N] = A[M,K](bf16) @ B[K,N](bf16). 256 thr = 8 waves; 128x128 tile, BK=32.
// Each wave: 32x64 sub-tile -> 2 A-frags, 4 B-frags, 8 WMMAs per k-step.
enum { EPI_BF16 = 0, EPI_F32_ADD = 1, EPI_GELU_BF16 = 2 };

#define AS_STR 40            /* 80B rows: 16B aligned, conflict-friendly */
#define BS_STR 34
#define B_OFF  (128*AS_STR)  /* u16 offset of BsT in smem */

__global__ __launch_bounds__(256) void gemm_bf16_wmma(const u16* __restrict__ A,
                                                      const u16* __restrict__ Bw,
                                                      void* __restrict__ Cout,
                                                      int M, int N, int Kd, int epi){
  __shared__ u16 smem[128*AS_STR + 128*BS_STR];
  const int tid  = threadIdx.x;
  const int wave = tid >> 5;
  const int lane = tid & 31;
  const int lh   = (lane >> 4) & 1;
  const int l15  = lane & 15;
  const int rowgrp = wave >> 1;     // 0..3 -> 32-row group
  const int colgrp = wave & 1;      // 0..1 -> 64-col group
  const int m0 = blockIdx.y * 128;
  const int n0 = blockIdx.x * 128;

  v8f acc[2][4];
  for (int ar = 0; ar < 2; ++ar)
    for (int t = 0; t < 4; ++t)
      for (int r = 0; r < 8; ++r) acc[ar][t][r] = 0.0f;

  for (int k0 = 0; k0 < Kd; k0 += 32){
    // --- async stage A: 128x32 (no transpose) via global_load_async_to_lds_b128
    #pragma unroll
    for (int p = 0; p < 2; ++p){
      const int r  = p*64 + (tid >> 2);
      const int c8 = (tid & 3) * 8;
      const u16* gsrc = A + (size_t)(m0 + r) * Kd + k0 + c8;
      async_ld_b128((u32)((r*AS_STR + c8) * 2), gsrc);
    }
    // --- stage B transposed: [k][n] -> BsT[n][k]
    #pragma unroll
    for (int p = 0; p < 2; ++p){
      const int kk = p*16 + (tid >> 4);
      const int nc = (tid & 15) * 8;
      const u16* src = Bw + (size_t)(k0 + kk) * N + n0 + nc;
      uint4 d = *(const uint4*)src;
      u32 w0[4] = {d.x, d.y, d.z, d.w};
      #pragma unroll
      for (int q = 0; q < 8; ++q)
        smem[B_OFF + (nc + q)*BS_STR + kk] = (u16)(w0[q >> 1] >> ((q & 1) * 16));
    }
    if (k0 + 32 < Kd)
      __builtin_prefetch(Bw + (size_t)(k0 + 32 + (tid >> 4)) * N + n0, 0, 1);
    async_wait0();
    __syncthreads();

    v16bf afr[2];
    #pragma unroll
    for (int ar = 0; ar < 2; ++ar){
      v8u au;
      const int arow = 32*rowgrp + 16*ar + l15;
      #pragma unroll
      for (int v = 0; v < 8; ++v){
        const int kb = ((v < 4) ? 2*v : 16 + 2*(v - 4)) + (lh ? 8 : 0);
        au[v] = *(const u32*)&smem[arow*AS_STR + kb];
      }
      afr[ar] = asbf(au);
    }
    #pragma unroll
    for (int t = 0; t < 4; ++t){
      v8u bu;
      const int brow = 64*colgrp + 16*t + l15;
      #pragma unroll
      for (int v = 0; v < 8; ++v){
        const int kb = ((v < 4) ? 2*v : 16 + 2*(v - 4)) + (lh ? 8 : 0);
        bu[v] = *(const u32*)&smem[B_OFF + brow*BS_STR + kb];
      }
      const v16bf bfr = asbf(bu);
      acc[0][t] = __builtin_amdgcn_wmma_f32_16x16x32_bf16(false, afr[0], false, bfr,
                                                          (short)0, acc[0][t], false, false);
      acc[1][t] = __builtin_amdgcn_wmma_f32_16x16x32_bf16(false, afr[1], false, bfr,
                                                          (short)0, acc[1][t], false, false);
    }
    __syncthreads();
  }

  #pragma unroll
  for (int ar = 0; ar < 2; ++ar){
    #pragma unroll
    for (int t = 0; t < 4; ++t){
      const int col = n0 + 64*colgrp + 16*t + l15;
      #pragma unroll
      for (int r = 0; r < 8; ++r){
        const int row = m0 + 32*rowgrp + 16*ar + r + (lh ? 8 : 0);
        const size_t off = (size_t)row * N + col;
        const float v = acc[ar][t][r];
        if (epi == EPI_BF16)         ((u16*)Cout)[off] = f2bf(v);
        else if (epi == EPI_F32_ADD) ((float*)Cout)[off] += v;
        else                         ((u16*)Cout)[off] = f2bf(gelu_f(v));
      }
    }
  }
}

// ---------------------------------------------------------------- flash attention
// grid (Lq/64, HEADS, B); block 128 (4 waves, 1 wave per 16-query tile).
// K/V tiles staged cooperatively in LDS (K via async b128, V transposed).
#define KT_STR 40
#define VT_OFF (32*KT_STR)                 /* u16 offsets in smem */
#define PT_OFF (VT_OFF + 64*34)
#define PT_WV  (16*34)

__global__ __launch_bounds__(128) void flash_attn(const u16* __restrict__ Qb,
                                                  const u16* __restrict__ Kb,
                                                  const u16* __restrict__ Vb,
                                                  u16* __restrict__ Ob,
                                                  const int* __restrict__ counts,
                                                  int Lq, int Lk, int useMask){
  __shared__ u16 smem[32*KT_STR + 64*34 + 4*PT_WV];
  const int tid  = threadIdx.x;
  const int lane = tid & 31;
  const int wave = tid >> 5;
  const int lh   = (lane >> 4) & 1;
  const int l15  = lane & 15;
  const int b  = blockIdx.z;
  const int h  = blockIdx.y;
  const int q0 = blockIdx.x * 64 + wave * 16;
  const int d0 = h * DH;

  // Q fragments: 16x64 as two K=32 A-frags
  v8u aq0u, aq1u;
  {
    const u16* qrow = Qb + ((size_t)b * Lq + q0 + l15) * DMODEL + d0;
    #pragma unroll
    for (int v = 0; v < 8; ++v){
      const int kb = ((v < 4) ? 2*v : 16 + 2*(v - 4)) + (lh ? 8 : 0);
      aq0u[v] = *(const u32*)(qrow + kb);
      aq1u[v] = *(const u32*)(qrow + 32 + kb);
    }
  }
  const v16bf aq0 = asbf(aq0u), aq1 = asbf(aq1u);

  float mrow[8], lrow[8];
  v8f o[4];
  for (int r = 0; r < 8; ++r){ mrow[r] = -1e30f; lrow[r] = 0.0f; }
  for (int t = 0; t < 4; ++t)
    for (int r = 0; r < 8; ++r) o[t][r] = 0.0f;

  const int cnt = useMask ? counts[b] : Lk;

  for (int j = 0; j < Lk; j += 32){
    // --- stage K tile 32x64 (async, row-major) and V tile transposed [d][key]
    #pragma unroll
    for (int p = 0; p < 2; ++p){
      const int row = (tid >> 3) + p*16;
      const int c8  = (tid & 7) * 8;
      const size_t gro = ((size_t)b * Lk + j + row) * DMODEL + d0 + c8;
      async_ld_b128((u32)((row*KT_STR + c8) * 2), Kb + gro);
      uint4 d = *(const uint4*)(Vb + gro);
      u32 w0[4] = {d.x, d.y, d.z, d.w};
      #pragma unroll
      for (int q = 0; q < 8; ++q)
        smem[VT_OFF + (c8 + q)*34 + row] = (u16)(w0[q >> 1] >> ((q & 1) * 16));
    }
    async_wait0();
    __syncthreads();

    // ---- scores: two 16x16 tiles (Q 16x64 @ K^T 64x16), K from LDS
    v8f s[2];
    #pragma unroll
    for (int kk = 0; kk < 2; ++kk){
      v8u b0u, b1u;
      const int krow = (16*kk + l15) * KT_STR;
      #pragma unroll
      for (int v = 0; v < 8; ++v){
        const int kb = ((v < 4) ? 2*v : 16 + 2*(v - 4)) + (lh ? 8 : 0);
        b0u[v] = *(const u32*)&smem[krow + kb];
        b1u[v] = *(const u32*)&smem[krow + 32 + kb];
      }
      v8f z;
      for (int r = 0; r < 8; ++r) z[r] = 0.0f;
      z = __builtin_amdgcn_wmma_f32_16x16x32_bf16(false, aq0, false, asbf(b0u), (short)0, z, false, false);
      z = __builtin_amdgcn_wmma_f32_16x16x32_bf16(false, aq1, false, asbf(b1u), (short)0, z, false, false);
      s[kk] = z;
    }
    // scale + key mask (column n = l15 in C layout)
    #pragma unroll
    for (int kk = 0; kk < 2; ++kk){
      const int key = j + 16*kk + l15;
      const bool valid = key < cnt;
      #pragma unroll
      for (int r = 0; r < 8; ++r){
        const float xv = s[kk][r] * 0.125f;
        s[kk][r] = valid ? xv : -1e9f;
      }
    }
    // ---- online softmax (rows at (r,half); cols across 16 lanes)
    #pragma unroll
    for (int r = 0; r < 8; ++r){
      float rm = fmaxf(s[0][r], s[1][r]);
      for (int dlt = 1; dlt < 16; dlt <<= 1)
        rm = fmaxf(rm, __shfl_xor(rm, dlt, 32));
      const float mnew = fmaxf(mrow[r], rm);
      const float corr = __expf(mrow[r] - mnew);
      const float p0 = __expf(s[0][r] - mnew);
      const float p1 = __expf(s[1][r] - mnew);
      const int m = r + (lh ? 8 : 0);
      smem[PT_OFF + wave*PT_WV + m*34 + l15]      = f2bf(p0);
      smem[PT_OFF + wave*PT_WV + m*34 + 16 + l15] = f2bf(p1);
      float rs = p0 + p1;
      for (int dlt = 1; dlt < 16; dlt <<= 1)
        rs += __shfl_xor(rs, dlt, 32);
      lrow[r] = lrow[r] * corr + rs;
      mrow[r] = mnew;
      o[0][r] *= corr; o[1][r] *= corr; o[2][r] *= corr; o[3][r] *= corr;
    }
    // P tile is per-wave; LDS ops are in-order within a wave -> no barrier needed
    v8u apu;
    #pragma unroll
    for (int v = 0; v < 8; ++v){
      const int kb = ((v < 4) ? 2*v : 16 + 2*(v - 4)) + (lh ? 8 : 0);
      apu[v] = *(const u32*)&smem[PT_OFF + wave*PT_WV + l15*34 + kb];
    }
    const v16bf ap = asbf(apu);
    #pragma unroll
    for (int t = 0; t < 4; ++t){
      v8u bvu;
      const int vrow = (16*t + l15) * 34;
      #pragma unroll
      for (int v = 0; v < 8; ++v){
        const int kb = ((v < 4) ? 2*v : 16 + 2*(v - 4)) + (lh ? 8 : 0);
        bvu[v] = *(const u32*)&smem[VT_OFF + vrow + kb];
      }
      o[t] = __builtin_amdgcn_wmma_f32_16x16x32_bf16(false, ap, false, asbf(bvu),
                                                     (short)0, o[t], false, false);
    }
    __syncthreads();   // protect K/V/P before next-iter staging
  }

  #pragma unroll
  for (int t = 0; t < 4; ++t){
    const int col = d0 + 16*t + l15;
    #pragma unroll
    for (int r = 0; r < 8; ++r){
      const int row = q0 + r + (lh ? 8 : 0);
      const float l = lrow[r];
      const float inv = (l > 0.0f) ? 1.0f / l : 0.0f;
      Ob[((size_t)b * Lq + row) * DMODEL + col] = f2bf(o[t][r] * inv);
    }
  }
}

// ---------------------------------------------------------------- host driver
extern "C" void kernel_launch(void* const* d_in, const int* in_sizes, int n_in,
                              void* d_out, int out_size, void* d_ws, size_t ws_size,
                              hipStream_t stream){
  const float* img   = (const float*)d_in[0];
  const float* hid   = (const float*)d_in[1];
  const int*   seq   = (const int*)  d_in[2];
  const float* wq_s  = (const float*)d_in[3];
  const float* wk_s  = (const float*)d_in[4];
  const float* wv_s  = (const float*)d_in[5];
  const float* wo_s  = (const float*)d_in[6];
  const float* wq_c  = (const float*)d_in[7];
  const float* wk_c  = (const float*)d_in[8];
  const float* wv_c  = (const float*)d_in[9];
  const float* wo_c  = (const float*)d_in[10];
  const float* w_ff1 = (const float*)d_in[11];
  const float* w_ff2 = (const float*)d_in[12];
  const float* ln1_s = (const float*)d_in[13];
  const float* ln1_b = (const float*)d_in[14];
  const float* ln2_s = (const float*)d_in[15];
  const float* ln2_b = (const float*)d_in[16];
  const float* ln3_s = (const float*)d_in[17];
  const float* ln3_b = (const float*)d_in[18];

  char* ws = (char*)d_ws;
  size_t off = 0;
  auto alloc = [&](size_t bytes) -> void* {
    void* p = ws + off;
    off += (bytes + 255) & ~(size_t)255;
    return p;
  };

  int*   pos    = (int*)  alloc(sizeof(int) * BDIM * SEQ);
  int*   counts = (int*)  alloc(sizeof(int) * BDIM);
  float* x      = (float*)alloc(sizeof(float) * (size_t)MQ * DMODEL);
  u16*   lnb    = (u16*)  alloc(sizeof(u16) * (size_t)MQ * DMODEL);
  u16*   qb     = (u16*)  alloc(sizeof(u16) * (size_t)MQ * DMODEL);
  u16*   kb     = (u16*)  alloc(sizeof(u16) * (size_t)MI * DMODEL);
  u16*   vb     = (u16*)  alloc(sizeof(u16) * (size_t)MI * DMODEL);
  u16*   aob    = (u16*)  alloc(sizeof(u16) * (size_t)MQ * DMODEL);
  u16*   gb     = (u16*)  alloc(sizeof(u16) * (size_t)MQ * FFDIM);
  u16*   imgb   = (u16*)  alloc(sizeof(u16) * (size_t)MI * DMODEL);
  u16*   wb     = (u16*)  alloc(sizeof(u16) * (size_t)(8*DMODEL*DMODEL + 2*DMODEL*FFDIM));

  const size_t WSZ = (size_t)DMODEL * DMODEL;
  u16* wqs_b = wb + 0*WSZ;  u16* wks_b = wb + 1*WSZ;
  u16* wvs_b = wb + 2*WSZ;  u16* wos_b = wb + 3*WSZ;
  u16* wqc_b = wb + 4*WSZ;  u16* wkc_b = wb + 5*WSZ;
  u16* wvc_b = wb + 6*WSZ;  u16* woc_b = wb + 7*WSZ;
  u16* ff1_b = wb + 8*WSZ;  u16* ff2_b = ff1_b + (size_t)DMODEL * FFDIM;

  auto cvt = [&](const float* s, u16* d, long n){
    int blocks = (int)((n + 256*8 - 1) / (256*8));
    convert_f32_bf16<<<blocks, 256, 0, stream>>>(s, d, n);
  };
  cvt(wq_s,  wqs_b, WSZ);  cvt(wk_s,  wks_b, WSZ);
  cvt(wv_s,  wvs_b, WSZ);  cvt(wo_s,  wos_b, WSZ);
  cvt(wq_c,  wqc_b, WSZ);  cvt(wk_c,  wkc_b, WSZ);
  cvt(wv_c,  wvc_b, WSZ);  cvt(wo_c,  woc_b, WSZ);
  cvt(w_ff1, ff1_b, (long)DMODEL * FFDIM);
  cvt(w_ff2, ff2_b, (long)FFDIM * DMODEL);
  cvt(img,   imgb,  (long)MI * DMODEL);

  // mask scan + zero-padded extract
  zero_f32<<<2048, 256, 0, stream>>>(x, (long)MQ * DMODEL);
  scan_kernel<<<BDIM, 256, 0, stream>>>(seq, pos, counts);
  extract_kernel<<<BDIM * SEQ, 256, 0, stream>>>(hid, pos, x);

  auto gemm = [&](const u16* A, const u16* Bm, void* C, int M, int N, int K, int epi){
    dim3 g(N / 128, M / 128);
    gemm_bf16_wmma<<<g, 256, 0, stream>>>(A, Bm, C, M, N, K, epi);
  };

  // ---- Block 1: self attention (pre-LN, key mask = run)
  ln_kernel<<<MQ, 256, 0, stream>>>(x, ln1_s, ln1_b, lnb);
  gemm(lnb, wqs_b, qb, MQ, DMODEL, DMODEL, EPI_BF16);
  gemm(lnb, wks_b, kb, MQ, DMODEL, DMODEL, EPI_BF16);
  gemm(lnb, wvs_b, vb, MQ, DMODEL, DMODEL, EPI_BF16);
  {
    dim3 g(CAP / 64, HEADS, BDIM);
    flash_attn<<<g, 128, 0, stream>>>(qb, kb, vb, aob, counts, CAP, CAP, 1);
  }
  gemm(aob, wos_b, x, MQ, DMODEL, DMODEL, EPI_F32_ADD);     // residual fused

  // ---- Block 2: cross attention to img (no mask)
  ln_kernel<<<MQ, 256, 0, stream>>>(x, ln2_s, ln2_b, lnb);
  gemm(lnb,  wqc_b, qb, MQ, DMODEL, DMODEL, EPI_BF16);
  gemm(imgb, wkc_b, kb, MI, DMODEL, DMODEL, EPI_BF16);
  gemm(imgb, wvc_b, vb, MI, DMODEL, DMODEL, EPI_BF16);
  {
    dim3 g(CAP / 64, HEADS, BDIM);
    flash_attn<<<g, 128, 0, stream>>>(qb, kb, vb, aob, counts, CAP, NIMG, 0);
  }
  gemm(aob, woc_b, x, MQ, DMODEL, DMODEL, EPI_F32_ADD);     // residual fused

  // ---- Block 3: FFN with fused GELU epilogue
  ln_kernel<<<MQ, 256, 0, stream>>>(x, ln3_s, ln3_b, lnb);
  gemm(lnb, ff1_b, gb, MQ, FFDIM, DMODEL, EPI_GELU_BF16);   // gelu fused
  gemm(gb,  ff2_b, x,  MQ, DMODEL, FFDIM, EPI_F32_ADD);     // residual fused

  // ---- outputs: hid_new, then x
  float* out_hid = (float*)d_out;
  float* out_x   = out_hid + (size_t)BDIM * SEQ * DMODEL;
  scatter_kernel<<<BDIM * SEQ, 256, 0, stream>>>(hid, x, pos, out_hid);
  copy_f32<<<2048, 256, 0, stream>>>(x, out_x, (long)MQ * DMODEL);
}
